// MicroVoxelSpatialEncoder_5540507812249
// MI455X (gfx1250) — matrix operand order repaired and checked by reading, not compiled, not run on hardware
//
#include <hip/hip_runtime.h>
#include <math.h>

// ---------------- problem constants ----------------
#define BB   2
#define NN   4096
#define DIN  32
#define DD   128
#define HH   4
#define KK   27
#define GXD  64
#define GYD  64
#define GZD  100
#define VV   (GXD*GYD*GZD)     // 409600
#define BN   (BB*NN)           // 8192

typedef __attribute__((ext_vector_type(16))) _Float16 v16h;
typedef __attribute__((ext_vector_type(8)))  float    v8f;

// ---------------- WMMA helpers (CDNA5 layouts, CDNA5 ISA) ----------------
__device__ __forceinline__ v8f wmma_f16(v16h a, v16h b, v8f c) {
  // (neg_a, A, neg_b, B, c_mod, C, reuse_a, reuse_b)
  return __builtin_amdgcn_wmma_f32_16x16x32_f16(false, a, false, b, (short)0, c, false, false);
}

// A tile 16(M)x32(K) f16 from row-major f32 (vectorized float4 loads; rows must be 16B aligned)
__device__ __forceinline__ v16h load_a_f32(const float* X, int ld, int lane, float scale) {
  int m  = lane & 15;
  int kh = (lane >> 4) ? 8 : 0;
  const float* base = X + (size_t)m * ld + kh;
  float4 x0 = *(const float4*)(base);
  float4 x1 = *(const float4*)(base + 4);
  float4 x2 = *(const float4*)(base + 16);
  float4 x3 = *(const float4*)(base + 20);
  v16h a;
  a[0]  = (_Float16)(x0.x * scale); a[1]  = (_Float16)(x0.y * scale);
  a[2]  = (_Float16)(x0.z * scale); a[3]  = (_Float16)(x0.w * scale);
  a[4]  = (_Float16)(x1.x * scale); a[5]  = (_Float16)(x1.y * scale);
  a[6]  = (_Float16)(x1.z * scale); a[7]  = (_Float16)(x1.w * scale);
  a[8]  = (_Float16)(x2.x * scale); a[9]  = (_Float16)(x2.y * scale);
  a[10] = (_Float16)(x2.z * scale); a[11] = (_Float16)(x2.w * scale);
  a[12] = (_Float16)(x3.x * scale); a[13] = (_Float16)(x3.y * scale);
  a[14] = (_Float16)(x3.z * scale); a[15] = (_Float16)(x3.w * scale);
  return a;
}

// B tile from pre-packed f16 array laid out as [kc][t][lane][16 halves]
__device__ __forceinline__ v16h load_b_packed(const _Float16* Wp, int kc, int t, int lane) {
  return *(const v16h*)(Wp + (size_t)(((kc * 8 + t) * 32 + lane) << 4));
}

// ---------------- weight pre-pack into WMMA B layout (f16) ----------------
__global__ void k_pack(const float* __restrict__ W, _Float16* __restrict__ out,
                       int ld, int trans, int kchunks) {
  int tid = blockIdx.x * blockDim.x + threadIdx.x;
  int total = kchunks << 12;           // kchunks*8*32*16
  if (tid >= total) return;
  int j    = tid & 15;
  int lane = (tid >> 4) & 31;
  int t    = (tid >> 9) & 7;
  int kc   = tid >> 12;
  int k = kc * 32 + ((lane >> 4) ? 8 : 0) + (j & 7) + ((j >> 3) ? 16 : 0);
  int n = t * 16 + (lane & 15);
  float v = trans ? W[(size_t)n * ld + k] : W[(size_t)k * ld + n];
  out[tid] = (_Float16)v;
}

// ---------------- kernel 0: init scratch ----------------
__global__ void k_init(int* cnt, int* slot, int* nocc, int* cntc, float* gsum) {
  int i = blockIdx.x * blockDim.x + threadIdx.x;
  int stride = gridDim.x * blockDim.x;
  for (int j = i; j < BB * VV; j += stride) { cnt[j] = 0; slot[j] = -1; }
  for (int j = i; j < BN * DD; j += stride) gsum[j] = 0.0f;
  for (int j = i; j < BN; j += stride) cntc[j] = 0;
  if (i < BB) nocc[i] = 0;
}

// ---------------- kernel 1: voxelize + per-voxel count ----------------
__global__ void k_points(const float* coords, int* vidx3, int* cnt) {
  int p = blockIdx.x * blockDim.x + threadIdx.x;
  if (p >= BN) return;
  int b = p >> 12;
  const float* c = coords + (size_t)p * 3;
  float nx = fminf(fmaxf(c[0] * (1.0f / 256.0f), 0.0f), 1.0f);
  float ny = fminf(fmaxf(c[1] * (1.0f / 256.0f), 0.0f), 1.0f);
  float nz = fminf(fmaxf(c[2], 0.0f), 1.0f);
  int vx = (int)(nx * 63.0f);
  int vy = (int)(ny * 63.0f);
  int vz = (int)(nz * 99.0f);
  vidx3[p * 3 + 0] = vx; vidx3[p * 3 + 1] = vy; vidx3[p * 3 + 2] = vz;
  int vid = (vx * GYD + vy) * GZD + vz;
  atomicAdd(&cnt[b * VV + vid], 1);
}

// ---------------- kernel 2: compact occupied voxels ----------------
__global__ void k_compact(const int* cnt, int* slot, int* nocc, int* cntc) {
  int i = blockIdx.x * blockDim.x + threadIdx.x;
  if (i >= BB * VV) return;
  int c = cnt[i];
  if (c > 0) {
    int b = i / VV;
    int s = atomicAdd(&nocc[b], 1);   // occupied <= N per batch
    slot[i] = s;
    cntc[b * NN + s] = c;
  }
}

// ------- kernel 3: feat = features@fW+fb ; q = feat@Wq^T+bq (WMMA; fW async-staged in LDS) -------
__global__ __launch_bounds__(32) void k_feat_q(const float* __restrict__ feats,
                                               const _Float16* __restrict__ fWp,
                                               const float* __restrict__ fb,
                                               const _Float16* __restrict__ qWp,
                                               const float* __restrict__ in_b,
                                               float* __restrict__ feat,
                                               float* __restrict__ qout) {
  __shared__ __attribute__((aligned(32))) float    tile[16][136];
  __shared__ __attribute__((aligned(32))) _Float16 fWl[8 * 32 * 16];   // 8KB packed fW
  int lane = threadIdx.x;
  int row0 = blockIdx.x * 16;
  int col = lane & 15;
  int mb  = (lane >> 4) ? 8 : 0;
  __builtin_prefetch(qWp, 0, 2);

  // async-stage packed fW into LDS (ASYNCcnt path, CDNA5 ISA)
  unsigned ldsbase = (unsigned)(size_t)(&fWl[0]);   // low 32 bits of LDS flat addr = LDS offset
#pragma unroll
  for (int i = 0; i < 16; ++i) {
    unsigned off = (unsigned)((i * 32 + lane) * 16);
    asm volatile("global_load_async_to_lds_b128 %0, %1, %2"
                 :: "v"(ldsbase + off), "v"(off), "s"(fWp) : "memory");
  }

  // overlap: A-tile load/convert while the async copy is in flight
  v16h a = load_a_f32(feats + (size_t)row0 * DIN, DIN, lane, 1.0f);

  asm volatile("s_wait_asynccnt 0" ::: "memory");

  // feat GEMM: K = 32 (single K-chunk), B tiles from LDS-staged packed fW
#pragma unroll
  for (int t = 0; t < 8; ++t) {
    v16h b = *(const v16h*)(fWl + ((t * 32 + lane) << 4));
    v8f c = {};
    c = wmma_f16(a, b, c);
    int nn = t * 16 + col;
    float bias = fb[nn];
#pragma unroll
    for (int r = 0; r < 8; ++r) {
      float v = c[r] + bias;
      feat[(size_t)(row0 + mb + r) * DD + nn] = v;
      tile[mb + r][nn] = v;
    }
  }
  __syncthreads();

  // q GEMM: K = 128 from LDS feat tile; B = packed Wq^T
  v16h aq[4];
#pragma unroll
  for (int kc = 0; kc < 4; ++kc) aq[kc] = load_a_f32(&tile[0][0] + kc * 32, 136, lane, 1.0f);
#pragma unroll
  for (int t = 0; t < 8; ++t) {
    v8f c = {};
#pragma unroll
    for (int kc = 0; kc < 4; ++kc)
      c = wmma_f16(aq[kc], load_b_packed(qWp, kc, t, lane), c);
    int nn = t * 16 + col;
    float bias = in_b[nn];
#pragma unroll
    for (int r = 0; r < 8; ++r)
      qout[(size_t)(row0 + mb + r) * DD + nn] = c[r] + bias;
  }
}

// ---------------- kernel 4: scatter-add feat into compacted gsum ----------------
__global__ void k_scatter(const float* feat, const int* vidx3, const int* slot, float* gsum) {
  int tid = blockIdx.x * blockDim.x + threadIdx.x;
  if (tid >= BN * 32) return;
  int p = tid >> 5, lane = tid & 31;
  int b = p >> 12;
  int vx = vidx3[p * 3], vy = vidx3[p * 3 + 1], vz = vidx3[p * 3 + 2];
  int vid = (vx * GYD + vy) * GZD + vz;
  int s = slot[b * VV + vid];
  size_t dst = (size_t)(b * NN + s) * DD;
  size_t src = (size_t)p * DD;
#pragma unroll
  for (int h = 0; h < 4; ++h)
    atomicAdd(&gsum[dst + h * 32 + lane], feat[src + h * 32 + lane]);
}

// ------- kernel 5: grid_feat = relu(gmean@aW+ab); Kg/Vg projections (WMMA, LDS-fused) -------
__global__ __launch_bounds__(32) void k_grid(const float* __restrict__ gsum,
                                             const int* __restrict__ cntc,
                                             const _Float16* __restrict__ aWp,
                                             const float* __restrict__ ab,
                                             const _Float16* __restrict__ kWp,
                                             const _Float16* __restrict__ vWp,
                                             float* __restrict__ Kg,
                                             float* __restrict__ Vg) {
  __shared__ __attribute__((aligned(32))) float tile[16][136];
  int lane = threadIdx.x;
  int row0 = blockIdx.x * 16;          // global compacted row (b*NN + s); 4096%16==0
  int col = lane & 15;
  int mb  = (lane >> 4) ? 8 : 0;
  __builtin_prefetch(aWp, 0, 2);

  int m = lane & 15;
  float scale = 1.0f / fmaxf((float)cntc[row0 + m], 1.0f);

  v16h a[4];
#pragma unroll
  for (int kc = 0; kc < 4; ++kc)
    a[kc] = load_a_f32(gsum + (size_t)row0 * DD + kc * 32, DD, lane, scale);

#pragma unroll
  for (int t = 0; t < 8; ++t) {
    v8f c = {};
#pragma unroll
    for (int kc = 0; kc < 4; ++kc)
      c = wmma_f16(a[kc], load_b_packed(aWp, kc, t, lane), c);
    int nn = t * 16 + col;
    float bias = ab[nn];
#pragma unroll
    for (int r = 0; r < 8; ++r) tile[mb + r][nn] = fmaxf(c[r] + bias, 0.0f);
  }
  __syncthreads();

  v16h ag[4];
#pragma unroll
  for (int kc = 0; kc < 4; ++kc) ag[kc] = load_a_f32(&tile[0][0] + kc * 32, 136, lane, 1.0f);

#pragma unroll
  for (int t = 0; t < 8; ++t) {
    v8f ck = {}, cv = {};
#pragma unroll
    for (int kc = 0; kc < 4; ++kc) {
      ck = wmma_f16(ag[kc], load_b_packed(kWp, kc, t, lane), ck);
      cv = wmma_f16(ag[kc], load_b_packed(vWp, kc, t, lane), cv);
    }
    int nn = t * 16 + col;
#pragma unroll
    for (int r = 0; r < 8; ++r) {
      size_t o = (size_t)(row0 + mb + r) * DD + nn;
      Kg[o] = ck[r];                 // biases folded into pos tables
      Vg[o] = cv[r];
    }
  }
}

// ---------------- kernel 6: pos tables  posk/posv[27][128] (k/v biases folded in) ----------------
__global__ __launch_bounds__(128) void k_pos(const float* pW1, const float* pb1,
                                             const float* pW2, const float* pb2,
                                             const float* in_w, const float* in_b,
                                             float* posk, float* posv) {
  __shared__ float hid[64];
  __shared__ float pos[128];
  int kk = blockIdx.x, t = threadIdx.x;
  float ox = (float)(kk / 9 - 1);
  float oy = (float)((kk / 3) % 3 - 1);
  float oz = (float)(kk % 3 - 1);
  if (t < 64) {
    float v = ox * pW1[t] + oy * pW1[64 + t] + oz * pW1[128 + t] + pb1[t];
    hid[t] = fmaxf(v, 0.0f);
  }
  __syncthreads();
  float pv = pb2[t];
  for (int j = 0; j < 64; ++j) pv += hid[j] * pW2[j * DD + t];
  pos[t] = pv;
  __syncthreads();
  float sk = in_b[DD + t], sv = in_b[2 * DD + t];
  const float* wk = in_w + (size_t)(DD + t) * DD;
  const float* wv = in_w + (size_t)(2 * DD + t) * DD;
  for (int e = 0; e < 128; ++e) { float pe = pos[e]; sk += pe * wk[e]; sv += pe * wv[e]; }
  posk[kk * DD + t] = sk;
  posv[kk * DD + t] = sv;
}

// ---------------- kernel 7: attention (one wave32 per point) ----------------
__global__ __launch_bounds__(32) void k_attn(const int* __restrict__ vidx3,
                                             const int* __restrict__ slot,
                                             const float* __restrict__ qbuf,
                                             const float* __restrict__ Kg,
                                             const float* __restrict__ Vg,
                                             const float* __restrict__ posk,
                                             const float* __restrict__ posv,
                                             float* __restrict__ attn,
                                             int* __restrict__ has) {
  int p = blockIdx.x;
  int lane = threadIdx.x;
  int b = p >> 12;
  int vx = vidx3[p * 3], vy = vidx3[p * 3 + 1], vz = vidx3[p * 3 + 2];

  float qv[4];
#pragma unroll
  for (int h = 0; h < 4; ++h) qv[h] = qbuf[(size_t)p * DD + h * 32 + lane];

  int myslot = -1;
  if (lane < KK) {
    int nx = vx + lane / 9 - 1;
    int ny = vy + (lane / 3) % 3 - 1;
    int nz = vz + lane % 3 - 1;
    if (nx >= 0 && nx < GXD && ny >= 0 && ny < GYD && nz >= 0 && nz < GZD)
      myslot = slot[b * VV + (nx * GYD + ny) * GZD + nz];
  }
  int myvalid = (myslot >= 0) ? 1 : 0;

  const float rscale = 0.17677669529663687f;  // 1/sqrt(32)
  float sc[4] = {-1e30f, -1e30f, -1e30f, -1e30f};
  for (int kk = 0; kk < KK; ++kk) {
    int val = __shfl(myvalid, kk, 32);
    if (!val) continue;                       // uniform branch
    int s = __shfl(myslot, kk, 32);
    const float* kr = Kg + (size_t)(b * NN + s) * DD;
    const float* pr = posk + kk * DD;
    float part[4];
#pragma unroll
    for (int h = 0; h < 4; ++h) part[h] = qv[h] * (kr[h * 32 + lane] + pr[h * 32 + lane]);
#pragma unroll
    for (int mm = 16; mm >= 1; mm >>= 1)
#pragma unroll
      for (int h = 0; h < 4; ++h) part[h] += __shfl_xor(part[h], mm, 32);
    if (lane == kk)
#pragma unroll
      for (int h = 0; h < 4; ++h) sc[h] = part[h] * rscale;
  }

  float prob[4];
#pragma unroll
  for (int h = 0; h < 4; ++h) {
    float mx = sc[h];
#pragma unroll
    for (int mm = 16; mm >= 1; mm >>= 1) mx = fmaxf(mx, __shfl_xor(mx, mm, 32));
    float e = myvalid ? __expf(sc[h] - mx) : 0.0f;
    float sm = e;
#pragma unroll
    for (int mm = 16; mm >= 1; mm >>= 1) sm += __shfl_xor(sm, mm, 32);
    prob[h] = e / fmaxf(sm, 1e-30f);
  }

  float acc[4] = {0.0f, 0.0f, 0.0f, 0.0f};
  for (int kk = 0; kk < KK; ++kk) {
    int val = __shfl(myvalid, kk, 32);
    if (!val) continue;
    int s = __shfl(myslot, kk, 32);
    float ph[4];
#pragma unroll
    for (int h = 0; h < 4; ++h) ph[h] = __shfl(prob[h], kk, 32);
    const float* vr = Vg + (size_t)(b * NN + s) * DD;
    const float* pr = posv + kk * DD;
#pragma unroll
    for (int h = 0; h < 4; ++h) acc[h] += ph[h] * (vr[h * 32 + lane] + pr[h * 32 + lane]);
  }
#pragma unroll
  for (int h = 0; h < 4; ++h) attn[(size_t)p * DD + h * 32 + lane] = acc[h];
  unsigned long long bal = __ballot(myvalid);
  if (lane == 0) has[p] = (bal != 0ull) ? 1 : 0;
}

// ---- kernel 8: out = LN(feat + has*(attn@out_w^T+out_b))*gamma+beta (WMMA + LN epilogue) ----
__global__ __launch_bounds__(32) void k_final(const float* __restrict__ attn,
                                              const float* __restrict__ featbuf,
                                              const _Float16* __restrict__ oWp,
                                              const float* __restrict__ out_b,
                                              const float* __restrict__ gamma,
                                              const float* __restrict__ beta,
                                              const int* __restrict__ has,
                                              float* __restrict__ out) {
  int lane = threadIdx.x;
  int row0 = blockIdx.x * 16;
  int col = lane & 15;
  int mb  = (lane >> 4) ? 8 : 0;

  v16h a[4];
#pragma unroll
  for (int kc = 0; kc < 4; ++kc)
    a[kc] = load_a_f32(attn + (size_t)row0 * DD + kc * 32, DD, lane, 1.0f);

  v8f c[8];
#pragma unroll
  for (int t = 0; t < 8; ++t) {
    c[t] = {};
#pragma unroll
    for (int kc = 0; kc < 4; ++kc)
      c[t] = wmma_f16(a[kc], load_b_packed(oWp, kc, t, lane), c[t]);
  }

#pragma unroll
  for (int r = 0; r < 8; ++r) {
    int tok = row0 + mb + r;
    float hf = has[tok] ? 1.0f : 0.0f;
    float x[8];
    float s1 = 0.0f;
#pragma unroll
    for (int t = 0; t < 8; ++t) {
      int nn = t * 16 + col;
      x[t] = featbuf[(size_t)tok * DD + nn] + hf * (c[t][r] + out_b[nn]);
      s1 += x[t];
    }
#pragma unroll
    for (int mm = 8; mm >= 1; mm >>= 1) s1 += __shfl_xor(s1, mm, 32);  // stays in 16-lane half
    float mu = s1 * (1.0f / 128.0f);
    float s2 = 0.0f;
#pragma unroll
    for (int t = 0; t < 8; ++t) { float d = x[t] - mu; s2 += d * d; }
#pragma unroll
    for (int mm = 8; mm >= 1; mm >>= 1) s2 += __shfl_xor(s2, mm, 32);
    float inv = rsqrtf(s2 * (1.0f / 128.0f) + 1e-5f);
#pragma unroll
    for (int t = 0; t < 8; ++t) {
      int nn = t * 16 + col;
      out[(size_t)tok * DD + nn] = gamma[nn] * (x[t] - mu) * inv + beta[nn];
    }
  }
}

// ---------------- launch ----------------
static inline size_t alignup(size_t x) { return (x + 255) & ~(size_t)255; }

extern "C" void kernel_launch(void* const* d_in, const int* in_sizes, int n_in,
                              void* d_out, int out_size, void* d_ws, size_t ws_size,
                              hipStream_t stream) {
  const float* features = (const float*)d_in[0];
  const float* coords   = (const float*)d_in[1];
  const float* fW       = (const float*)d_in[2];
  const float* fb       = (const float*)d_in[3];
  const float* aW       = (const float*)d_in[4];
  const float* ab       = (const float*)d_in[5];
  const float* pW1      = (const float*)d_in[6];
  const float* pb1      = (const float*)d_in[7];
  const float* pW2      = (const float*)d_in[8];
  const float* pb2      = (const float*)d_in[9];
  const float* in_w     = (const float*)d_in[10];
  const float* in_b     = (const float*)d_in[11];
  const float* out_w    = (const float*)d_in[12];
  const float* out_b    = (const float*)d_in[13];
  const float* gamma    = (const float*)d_in[14];
  const float* beta     = (const float*)d_in[15];
  float* out = (float*)d_out;

  char* ws = (char*)d_ws;
  size_t off = 0;
  auto take = [&](size_t bytes) { char* p = ws + off; off += alignup(bytes); return p; };
  float* feat  = (float*)take(sizeof(float) * BN * DD);
  float* qbuf  = (float*)take(sizeof(float) * BN * DD);
  float* gsum  = (float*)take(sizeof(float) * BN * DD);
  float* Kg    = (float*)take(sizeof(float) * BN * DD);
  float* Vg    = (float*)take(sizeof(float) * BN * DD);
  float* attn  = (float*)take(sizeof(float) * BN * DD);
  int*   vidx3 = (int*)take(sizeof(int) * BN * 3);
  int*   slot  = (int*)take(sizeof(int) * BB * VV);
  int*   cnt   = (int*)take(sizeof(int) * BB * VV);
  int*   cntc  = (int*)take(sizeof(int) * BN);
  int*   nocc  = (int*)take(sizeof(int) * 64);
  int*   has   = (int*)take(sizeof(int) * BN);
  float* posk  = (float*)take(sizeof(float) * KK * DD);
  float* posv  = (float*)take(sizeof(float) * KK * DD);
  _Float16* fWp = (_Float16*)take(sizeof(_Float16) * 1 * 4096);
  _Float16* qWp = (_Float16*)take(sizeof(_Float16) * 4 * 4096);
  _Float16* kWp = (_Float16*)take(sizeof(_Float16) * 4 * 4096);
  _Float16* vWp = (_Float16*)take(sizeof(_Float16) * 4 * 4096);
  _Float16* aWp = (_Float16*)take(sizeof(_Float16) * 4 * 4096);
  _Float16* oWp = (_Float16*)take(sizeof(_Float16) * 4 * 4096);
  (void)ws_size; (void)in_sizes; (void)n_in; (void)out_size;

  k_pack<<<16, 256, 0, stream>>>(fW, fWp, DD, 0, 1);
  k_pack<<<64, 256, 0, stream>>>(in_w, qWp, DD, 1, 4);
  k_pack<<<64, 256, 0, stream>>>(in_w + DD * DD, kWp, DD, 1, 4);
  k_pack<<<64, 256, 0, stream>>>(in_w + 2 * DD * DD, vWp, DD, 1, 4);
  k_pack<<<64, 256, 0, stream>>>(aW, aWp, DD, 0, 4);
  k_pack<<<64, 256, 0, stream>>>(out_w, oWp, DD, 1, 4);

  k_init   <<<1024, 256, 0, stream>>>(cnt, slot, nocc, cntc, gsum);
  k_points <<<BN / 256, 256, 0, stream>>>(coords, vidx3, cnt);
  k_compact<<<(BB * VV) / 256, 256, 0, stream>>>(cnt, slot, nocc, cntc);
  k_feat_q <<<BN / 16, 32, 0, stream>>>(features, fWp, fb, qWp, in_b, feat, qbuf);
  k_scatter<<<(BN * 32) / 256, 256, 0, stream>>>(feat, vidx3, slot, gsum);
  k_grid   <<<BN / 16, 32, 0, stream>>>(gsum, cntc, aWp, ab, kWp, vWp, Kg, Vg);
  k_pos    <<<KK, 128, 0, stream>>>(pW1, pb1, pW2, pb2, in_w, in_b, posk, posv);
  k_attn   <<<BN, 32, 0, stream>>>(vidx3, slot, qbuf, Kg, Vg, posk, posv, attn, has);
  k_final  <<<BN / 16, 32, 0, stream>>>(attn, feat, oWp, out_b, gamma, beta, has, out);
}
